// SINE_17497696764633
// MI455X (gfx1250) — compile-verified, matrix operand order checked by reading, not run on hardware
//
#include <hip/hip_runtime.h>
#include <hip/hip_bf16.h>
#include <math.h>

// ---------------- problem constants (match reference) ----------------
#define B_    1024
#define S_    50
#define H_    128
#define KP_   4
#define L_    500
#define NI_   100000
#define BS_   (B_ * S_)
#define EPS_  1e-5f
#define NTILES_ (NI_ / 16)      // 6250 exactly (100000 = 16*6250)

typedef __attribute__((ext_vector_type(16))) __bf16 v16bf;
typedef __attribute__((ext_vector_type(8)))  float  v8f;

// ---------------- WMMA helpers ----------------
__device__ __forceinline__ v8f wmma_bf16(v16bf a, v16bf b, v8f c) {
  return __builtin_amdgcn_wmma_f32_16x16x32_bf16(false, a, false, b, (short)0, c,
                                                 false, false);
}

// 16x32 bf16 A-fragment from a row-major f32 matrix with row stride 128.
// ISA layout: lane L holds row m=L&15; VGPR v element e -> k =
//   ((v&4)?16:0) + (L>>4)*8 + (v&3)*2 + e   (plus kstep*32)
__device__ __forceinline__ v16bf load_a_f32(const float* __restrict__ X,
                                            int row0, int ks, int lane) {
  const float* xr = X + (size_t)(row0 + (lane & 15)) * H_ + ks * 32;
  int half8 = (lane >> 4) * 8;
  v16bf a;
#pragma unroll
  for (int v = 0; v < 8; ++v) {
    int k = ((v & 4) ? 16 : 0) + half8 + (v & 3) * 2;
    a[2 * v]     = (__bf16)xr[k];
    a[2 * v + 1] = (__bf16)xr[k + 1];
  }
  return a;
}

__device__ __forceinline__ v16bf load_a_bf16(const __bf16* __restrict__ X,
                                             int row0, int ks, int lane) {
  const __bf16* xr = X + (size_t)(row0 + (lane & 15)) * H_ + ks * 32;
  int half8 = (lane >> 4) * 8;
  v16bf a;
#pragma unroll
  for (int v = 0; v < 8; ++v) {
    int k = ((v & 4) ? 16 : 0) + half8 + (v & 3) * 2;
    a[2 * v]     = xr[k];
    a[2 * v + 1] = xr[k + 1];
  }
  return a;
}

// B-fragment from a K-contiguous bf16 source: src[n][k] (n = output column).
// Covers B[k][n] for both W^T buffers and item_emb (scores GEMM), and the
// per-lane addresses are contiguous -> compiler emits global_load_b128 pairs.
__device__ __forceinline__ v16bf load_bT(const __bf16* __restrict__ src,
                                         int nbase, int ks, int lane) {
  const __bf16* r = src + (size_t)(nbase + (lane & 15)) * H_ + ks * 32;
  int half8 = (lane >> 4) * 8;
  v16bf b;
#pragma unroll
  for (int v = 0; v < 8; ++v) {
    int k = ((v & 4) ? 16 : 0) + half8 + (v & 3) * 2;
    b[2 * v]     = r[k];
    b[2 * v + 1] = r[k + 1];
  }
  return b;
}

// sum over the 16 lanes of each half-wave (wave32, xor masks 1,2,4,8)
__device__ __forceinline__ float half16_sum(float v) {
  v += __shfl_xor(v, 1, 32);
  v += __shfl_xor(v, 2, 32);
  v += __shfl_xor(v, 4, 32);
  v += __shfl_xor(v, 8, 32);
  return v;
}

// ---------------- block reductions (blockDim.x == 128) ----------------
__device__ __forceinline__ float blk_sum128(float v, float* red) {
  int t = threadIdx.x;
  red[t] = v; __syncthreads();
#pragma unroll
  for (int off = 64; off > 0; off >>= 1) {
    if (t < off) red[t] += red[t + off];
    __syncthreads();
  }
  float r = red[0]; __syncthreads();
  return r;
}
__device__ __forceinline__ float blk_max128(float v, float* red) {
  int t = threadIdx.x;
  red[t] = v; __syncthreads();
#pragma unroll
  for (int off = 64; off > 0; off >>= 1) {
    if (t < off) red[t] = fmaxf(red[t], red[t + off]);
    __syncthreads();
  }
  float r = red[0]; __syncthreads();
  return r;
}

// W[128,128] f32 row-major  ->  Wt[n][k] bf16 (K-contiguous per column)
__global__ void k_w_to_bf16T(const float* __restrict__ W, __bf16* __restrict__ Wt) {
  int n = blockIdx.x;    // 128 blocks
  int k = threadIdx.x;   // 128 threads: contiguous writes along k
  Wt[(size_t)n * H_ + k] = (__bf16)W[(size_t)k * H_ + n];
}

// =====================================================================
// Generic WMMA GEMM: Y = X[M,128] @ W[128,128], one wave per 16-row strip,
// wave computes the FULL 128-wide strip (8 col tiles x 4 k-steps = 32 WMMAs)
// so epilogues needing whole rows (layernorm / tanh-dot) run in-register.
// W passed pre-transposed bf16 (Wt[n][k]) -> contiguous b128 fragment loads.
// MODE 0: out[m,p] = sum_n tanh(Y[m,n]) * V[n,p]           (P = 1 or 4)
// MODE 1: out[m,:] = layernorm(Y[m,:]) * g + b             ([M,128])
// =====================================================================
template <int MODE, int P>
__global__ void gemm_h128(const float* __restrict__ X, const __bf16* __restrict__ Wt,
                          const float* __restrict__ V, const float* __restrict__ lng,
                          const float* __restrict__ lnb, float* __restrict__ out,
                          int mtiles) {
  int wave = threadIdx.x >> 5;
  int lane = threadIdx.x & 31;
  int strip = blockIdx.x * (blockDim.x >> 5) + wave;
  if (strip >= mtiles) return;           // wave-uniform
  int row0 = strip * 16;
  int n15 = lane & 15;
  int half = lane >> 4;

  v16bf afr[4];
#pragma unroll
  for (int ks = 0; ks < 4; ++ks) afr[ks] = load_a_f32(X, row0, ks, lane);

  v8f acc[8];
#pragma unroll
  for (int ct = 0; ct < 8; ++ct) {
    v8f c;
#pragma unroll
    for (int i = 0; i < 8; ++i) c[i] = 0.f;
#pragma unroll
    for (int ks = 0; ks < 4; ++ks) {
      v16bf bfr = load_bT(Wt, ct * 16, ks, lane);
      c = wmma_bf16(afr[ks], bfr, c);
    }
    acc[ct] = c;
  }

  if (MODE == 0) {
    float sum[8][P];
#pragma unroll
    for (int r = 0; r < 8; ++r)
#pragma unroll
      for (int p = 0; p < P; ++p) sum[r][p] = 0.f;
#pragma unroll
    for (int ct = 0; ct < 8; ++ct) {
      int n = ct * 16 + n15;
      float vv[P];
#pragma unroll
      for (int p = 0; p < P; ++p) vv[p] = V[n * P + p];
#pragma unroll
      for (int r = 0; r < 8; ++r) {
        float tv = tanhf(acc[ct][r]);
#pragma unroll
        for (int p = 0; p < P; ++p) sum[r][p] += tv * vv[p];
      }
    }
#pragma unroll
    for (int r = 0; r < 8; ++r)
#pragma unroll
      for (int p = 0; p < P; ++p) {
        float red = half16_sum(sum[r][p]);    // full 128-col dot for row r+8*half
        if (n15 == 0) {
          int m = row0 + r + 8 * half;
          out[(size_t)m * P + p] = red;
        }
      }
  } else {
#pragma unroll
    for (int r = 0; r < 8; ++r) {
      float s1 = 0.f, s2 = 0.f;
#pragma unroll
      for (int ct = 0; ct < 8; ++ct) {
        float y = acc[ct][r];
        s1 += y; s2 += y * y;
      }
      s1 = half16_sum(s1);
      s2 = half16_sum(s2);
      float mean = s1 * (1.f / H_);
      float var  = s2 * (1.f / H_) - mean * mean;
      float inv  = rsqrtf(var + EPS_);
      int m = row0 + r + 8 * half;
#pragma unroll
      for (int ct = 0; ct < 8; ++ct) {
        int n = ct * 16 + n15;
        out[(size_t)m * H_ + n] = (acc[ct][r] - mean) * inv * lng[n] + lnb[n];
      }
    }
  }
}

// =====================================================================
// scores[1024,100000] = final[1024,128] @ item_emb^T     (bf16 WMMA)
// N = 100000 = 16 * 6250 -> no N bounds checks. 8 waves/block, wave = n-tile.
// Memory-bound on the 410MB f32 store. item_emb (25.6MB bf16) is re-read 64x
// (once per m-tile) so it must stay L2-resident: stream the output with
// NON-TEMPORAL stores so the write traffic doesn't evict it.
// =====================================================================
__global__ void gemm_scores(const __bf16* __restrict__ A, const __bf16* __restrict__ Bm,
                            float* __restrict__ out) {
  int wave = threadIdx.x >> 5;
  int lane = threadIdx.x & 31;
  int ntile = blockIdx.x * 8 + wave;
  if (ntile >= NTILES_) return;          // wave-uniform
  int mtile = blockIdx.y;
  int n15 = lane & 15, half = lane >> 4;
  int n = ntile * 16 + n15;

  if (ntile + 8 < NTILES_)               // hint next block-column of items
    __builtin_prefetch(Bm + (size_t)(n + 128) * H_, 0, 1);

  v8f c;
#pragma unroll
  for (int i = 0; i < 8; ++i) c[i] = 0.f;
#pragma unroll
  for (int ks = 0; ks < 4; ++ks) {
    v16bf a = load_a_bf16(A, mtile * 16, ks, lane);
    v16bf b = load_bT(Bm, ntile * 16, ks, lane);
    c = wmma_bf16(a, b, c);
  }
#pragma unroll
  for (int r = 0; r < 8; ++r) {
    int m = mtile * 16 + r + 8 * half;
    __builtin_nontemporal_store(c[r], &out[(size_t)m * NI_ + n]);
  }
}

// ---------------- elementwise / small kernels (blockDim = 128) ----------------
__global__ void k_cvt_bf16(const float* __restrict__ x, __bf16* __restrict__ y,
                           long long n) {
  long long i = (long long)blockIdx.x * blockDim.x + threadIdx.x;
  if (i < n) y[i] = (__bf16)x[i];
}

__global__ void k_gather(const float* __restrict__ item, const float* __restrict__ pos,
                         const int* __restrict__ seq, float* __restrict__ ie,
                         float* __restrict__ emb, float* __restrict__ mask) {
  int row = blockIdx.x;                 // (b,s) flat, BS_ rows
  int h = threadIdx.x;
  int s = row % S_;
  int it = seq[row];
  float v = item[(size_t)it * H_ + h];
  ie[(size_t)row * H_ + h] = v;
  int position = S_ - s;                // SEQ - arange(SEQ)
  emb[(size_t)row * H_ + h] = v + pos[(size_t)position * H_ + h];
  if (h == 0) mask[row] = (it != 0) ? 1.f : 0.f;
}

__global__ void k_vcv(const float* __restrict__ sa, const float* __restrict__ mask,
                      const float* __restrict__ ie, float* __restrict__ vcv) {
  int b = blockIdx.x, t = threadIdx.x;
  __shared__ float red[128];
  __shared__ float al[S_];
  float x = (t < S_) ? sa[b * S_ + t] - (1.f - mask[b * S_ + t]) * 10000.f : -3e38f;
  float mx = blk_max128(x, red);
  float e = (t < S_) ? __expf(x - mx) : 0.f;
  float s = blk_sum128(e, red);
  if (t < S_) al[t] = e / s;
  __syncthreads();
  float acc = 0.f;
  for (int i = 0; i < S_; ++i) acc += al[i] * ie[((size_t)b * S_ + i) * H_ + t];
  vcv[(size_t)b * H_ + t] = acc;
}

__global__ void k_protoscore(const float* __restrict__ vcv, const float* __restrict__ proto,
                             float* __restrict__ sprot) {
  int b = blockIdx.x, t = threadIdx.x;
  __shared__ float v[H_];
  v[t] = vcv[(size_t)b * H_ + t];
  __syncthreads();
  for (int l = t; l < L_; l += 128) {
    float a = 0.f;
    for (int h = 0; h < H_; ++h) a += v[h] * proto[(size_t)l * H_ + h];
    sprot[(size_t)b * L_ + l] = a;
  }
}

__global__ void k_topk(const float* __restrict__ sprot, const float* __restrict__ proto,
                       const float* __restrict__ g2, const float* __restrict__ b2,
                       float* __restrict__ cp, float* __restrict__ cpln) {
  int b = blockIdx.x, t = threadIdx.x;
  __shared__ float sc[512];
  __shared__ float bv[128];
  __shared__ int bi[128];
  __shared__ float red[128];
  __shared__ int ridx[KP_];
  __shared__ float rval[KP_];
  for (int l = t; l < 512; l += 128)
    sc[l] = (l < L_) ? sprot[(size_t)b * L_ + l] : -3e38f;
  __syncthreads();
  for (int kk = 0; kk < KP_; ++kk) {
    float best = -3e38f; int besti = 511;
    for (int l = t; l < 512; l += 128) {
      float v = sc[l];
      if (v > best) { best = v; besti = l; }
    }
    bv[t] = best; bi[t] = besti;
    __syncthreads();
    for (int off = 64; off > 0; off >>= 1) {
      if (t < off) {
        if (bv[t + off] > bv[t] || (bv[t + off] == bv[t] && bi[t + off] < bi[t])) {
          bv[t] = bv[t + off]; bi[t] = bi[t + off];
        }
      }
      __syncthreads();
    }
    if (t == 0) { ridx[kk] = bi[0]; rval[kk] = bv[0]; sc[bi[0]] = -3e38f; }
    __syncthreads();
  }
  for (int kk = 0; kk < KP_; ++kk) {
    float sg = 1.f / (1.f + __expf(-rval[kk]));            // sigmoid(chosen_s)
    float val = sg * proto[(size_t)ridx[kk] * H_ + t];
    float mean = blk_sum128(val, red) * (1.f / H_);
    float d = val - mean;
    float var = blk_sum128(d * d, red) * (1.f / H_);
    size_t o = ((size_t)b * KP_ + kk) * H_ + t;
    cp[o] = val;
    cpln[o] = d * rsqrtf(var + EPS_) * g2[t] + b2[t];
  }
}

__global__ void k_pkt(const float* __restrict__ q, const float* __restrict__ cpln,
                      const float* __restrict__ mask, float* __restrict__ pkt) {
  int b = blockIdx.x, t = threadIdx.x;
  __shared__ float cps[KP_ * H_];
  for (int i = t; i < KP_ * H_; i += 128) cps[i] = cpln[(size_t)b * KP_ * H_ + i];
  __syncthreads();
  if (t < S_) {
    const float* qr = q + ((size_t)b * S_ + t) * H_;
    float sc[KP_];
#pragma unroll
    for (int k = 0; k < KP_; ++k) {
      float a = 0.f;
      for (int h = 0; h < H_; ++h) a += qr[h] * cps[k * H_ + h];
      sc[k] = a;
    }
    float mx = fmaxf(fmaxf(sc[0], sc[1]), fmaxf(sc[2], sc[3]));
    float e[KP_], ssum = 0.f;
#pragma unroll
    for (int k = 0; k < KP_; ++k) { e[k] = __expf(sc[k] - mx); ssum += e[k]; }
    float mk = mask[b * S_ + t];
#pragma unroll
    for (int k = 0; k < KP_; ++k)
      pkt[((size_t)b * S_ + t) * KP_ + k] = e[k] / ssum * mk;
  }
}

__global__ void k_ak(const float* __restrict__ aksc, const float* __restrict__ mask,
                     float* __restrict__ ak) {
  int bk = blockIdx.x;
  int b = bk >> 2, k = bk & 3;
  int t = threadIdx.x;
  __shared__ float red[128];
  float x = (t < S_) ? aksc[((size_t)b * S_ + t) * KP_ + k] : -3e38f;
  float mx = blk_max128(x, red);                // softmax over S (unmasked, per ref)
  float e = (t < S_) ? __expf(x - mx) : 0.f;
  float s = blk_sum128(e, red);
  if (t < S_) ak[((size_t)b * S_ + t) * KP_ + k] = e / s * mask[b * S_ + t];
}

__global__ void k_interesting(const float* __restrict__ pkt, const float* __restrict__ ak,
                              const float* __restrict__ emb, const float* __restrict__ g3,
                              const float* __restrict__ b3, float* __restrict__ inter) {
  int bk = blockIdx.x;
  int b = bk >> 2, k = bk & 3;
  int t = threadIdx.x;
  __shared__ float p[S_];
  __shared__ float red[128];
  if (t < S_) {
    size_t o = ((size_t)b * S_ + t) * KP_ + k;
    p[t] = pkt[o] * ak[o];
  }
  __syncthreads();
  float acc = 0.f;
  for (int s = 0; s < S_; ++s) acc += p[s] * emb[((size_t)b * S_ + s) * H_ + t];
  float mean = blk_sum128(acc, red) * (1.f / H_);
  float d = acc - mean;
  float var = blk_sum128(d * d, red) * (1.f / H_);
  inter[((size_t)b * KP_ + k) * H_ + t] = d * rsqrtf(var + EPS_) * g3[t] + b3[t];
}

__global__ void k_xu(const float* __restrict__ pkt, const float* __restrict__ cp,
                     float* __restrict__ xu) {
  int b = blockIdx.x, t = threadIdx.x;
  __shared__ float cps[KP_ * H_];
  __shared__ float pk[S_ * KP_];
  for (int i = t; i < KP_ * H_; i += 128) cps[i] = cp[(size_t)b * KP_ * H_ + i];
  for (int i = t; i < S_ * KP_; i += 128) pk[i] = pkt[(size_t)b * S_ * KP_ + i];
  __syncthreads();
  for (int s = 0; s < S_; ++s) {
    float a = 0.f;
#pragma unroll
    for (int k = 0; k < KP_; ++k) a += pk[s * KP_ + k] * cps[k * H_ + t];
    xu[((size_t)b * S_ + s) * H_ + t] = a;
  }
}

__global__ void k_capt(const float* __restrict__ csc, const float* __restrict__ mask,
                       const float* __restrict__ xu, const float* __restrict__ g4,
                       const float* __restrict__ b4, float* __restrict__ capt) {
  int b = blockIdx.x, t = threadIdx.x;
  __shared__ float red[128];
  __shared__ float cw[S_];
  float x = (t < S_) ? csc[b * S_ + t] - (1.f - mask[b * S_ + t]) * 10000.f : -3e38f;
  float mx = blk_max128(x, red);
  float e = (t < S_) ? __expf(x - mx) : 0.f;
  float s = blk_sum128(e, red);
  if (t < S_) cw[t] = e / s;
  __syncthreads();
  float acc = 0.f;
  for (int i = 0; i < S_; ++i) acc += cw[i] * xu[((size_t)b * S_ + i) * H_ + t];
  float mean = blk_sum128(acc, red) * (1.f / H_);
  float d = acc - mean;
  float var = blk_sum128(d * d, red) * (1.f / H_);
  capt[(size_t)b * H_ + t] = d * rsqrtf(var + EPS_) * g4[t] + b4[t];
}

__global__ void k_final(const float* __restrict__ capt, const float* __restrict__ inter,
                        float* __restrict__ finalf, __bf16* __restrict__ finalb) {
  int b = blockIdx.x, t = threadIdx.x;
  __shared__ float red[128];
  __shared__ float es[KP_];
  float c = capt[(size_t)b * H_ + t];
  for (int k = 0; k < KP_; ++k) {
    float s = blk_sum128(c * inter[((size_t)b * KP_ + k) * H_ + t], red);
    if (t == 0) es[k] = s;
    __syncthreads();
  }
  float mx = fmaxf(fmaxf(es[0], es[1]), fmaxf(es[2], es[3]));
  float ek[KP_], ssum = 0.f;
#pragma unroll
  for (int k = 0; k < KP_; ++k) { ek[k] = __expf((es[k] - mx) * 10.f); ssum += ek[k]; }  // /TAO
  float acc = 0.f;
#pragma unroll
  for (int k = 0; k < KP_; ++k) acc += ek[k] * inter[((size_t)b * KP_ + k) * H_ + t];
  acc /= ssum;
  finalf[(size_t)b * H_ + t] = acc;
  finalb[(size_t)b * H_ + t] = (__bf16)acc;
}

// ---------------- prototype covariance regularizer ----------------
__global__ void k_pmean(const float* __restrict__ proto, float* __restrict__ pmean,
                        float* __restrict__ accum) {
  int t = threadIdx.x;
  float s = 0.f;
  for (int l = 0; l < L_; ++l) s += proto[(size_t)l * H_ + t];
  pmean[t] = s * (1.f / L_);
  if (t < 2) accum[t] = 0.f;            // re-zero every launch (graph replay safe)
}

__global__ void k_gram(const float* __restrict__ proto, const float* __restrict__ pmean,
                       float* __restrict__ accum) {
  int i = blockIdx.x, t = threadIdx.x;
  __shared__ float ci[H_];
  __shared__ float red[128];
  ci[t] = proto[(size_t)i * H_ + t] - pmean[t];
  __syncthreads();
  float ss = 0.f, sd = 0.f;
  for (int j = t; j < L_; j += 128) {
    float d = 0.f;
    for (int h = 0; h < H_; ++h) d += ci[h] * (proto[(size_t)j * H_ + h] - pmean[h]);
    d *= (1.f / H_);
    ss += d * d;
    if (j == i) sd = d * d;
  }
  float tss = blk_sum128(ss, red);
  float tsd = blk_sum128(sd, red);
  if (t == 0) { atomicAdd(&accum[0], tss); atomicAdd(&accum[1], tsd); }
}

__global__ void k_fin(const float* __restrict__ accum, float* __restrict__ outp) {
  if (threadIdx.x == 0 && blockIdx.x == 0)
    outp[0] = 0.1f * 0.5f * (sqrtf(accum[0]) - sqrtf(accum[1]));  // LAMBDA*(||M||-||diag||)/2
}

// =====================================================================
extern "C" void kernel_launch(void* const* d_in, const int* in_sizes, int n_in,
                              void* d_out, int out_size, void* d_ws, size_t ws_size,
                              hipStream_t stream) {
  const float* item  = (const float*)d_in[0];
  const float* pos   = (const float*)d_in[1];
  const float* proto = (const float*)d_in[2];
  const float* w1 = (const float*)d_in[3];
  const float* w2 = (const float*)d_in[4];
  const float* w3 = (const float*)d_in[5];
  const float* w4 = (const float*)d_in[6];
  const float* w5 = (const float*)d_in[7];
  const float* w6 = (const float*)d_in[8];
  const float* w7 = (const float*)d_in[9];
  const float* ln1g = (const float*)d_in[10];
  const float* ln1b = (const float*)d_in[11];
  const float* ln2g = (const float*)d_in[12];
  const float* ln2b = (const float*)d_in[13];
  const float* ln3g = (const float*)d_in[14];
  const float* ln3b = (const float*)d_in[15];
  const float* ln4g = (const float*)d_in[16];
  const float* ln4b = (const float*)d_in[17];
  const int*   seq  = (const int*)d_in[18];

  float* scores = (float*)d_out;
  float* out_scalar = scores + (size_t)B_ * NI_;

  // ---- workspace carve-up (256B aligned slabs) ----
  char* wp = (char*)d_ws;
  auto alloc = [&](size_t nfloats) -> float* {
    float* p = (float*)wp;
    wp += ((nfloats * sizeof(float)) + 255) & ~(size_t)255;
    return p;
  };
  float* ie    = alloc((size_t)BS_ * H_);
  float* emb   = alloc((size_t)BS_ * H_);
  float* q     = alloc((size_t)BS_ * H_);       // layernorm(ie @ w3)
  float* xu    = alloc((size_t)BS_ * H_);
  float* mask  = alloc(BS_);
  float* sa    = alloc(BS_);
  float* vcv   = alloc((size_t)B_ * H_);
  float* sprot = alloc((size_t)B_ * L_);
  float* cp    = alloc((size_t)B_ * KP_ * H_);  // sigmoid(s)*proto[rank]
  float* cpln  = alloc((size_t)B_ * KP_ * H_);  // ln2 of the above
  float* pkt   = alloc((size_t)BS_ * KP_);
  float* aksc  = alloc((size_t)BS_ * KP_);
  float* ak    = alloc((size_t)BS_ * KP_);
  float* inter = alloc((size_t)B_ * KP_ * H_);
  float* csc   = alloc(BS_);
  float* capt  = alloc((size_t)B_ * H_);
  float* finalf = alloc((size_t)B_ * H_);
  __bf16* finalb = (__bf16*)alloc((size_t)B_ * H_ / 2 + 64);
  __bf16* itemb  = (__bf16*)alloc((size_t)NI_ * H_ / 2);
  __bf16* wt1 = (__bf16*)alloc((size_t)H_ * H_ / 2);  // transposed bf16 weights
  __bf16* wt3 = (__bf16*)alloc((size_t)H_ * H_ / 2);
  __bf16* wt4 = (__bf16*)alloc((size_t)H_ * H_ / 2);
  __bf16* wt6 = (__bf16*)alloc((size_t)H_ * H_ / 2);
  float* pmean = alloc(H_);
  float* accum = alloc(2);
  (void)ws_size; (void)in_sizes; (void)n_in; (void)out_size;

  const int mstrips = BS_ / 16;                 // 3200
  dim3 blk128(128), blk256(256);

  // independent: precision/layout conversions + prototype regularizer chain
  {
    long long n = (long long)NI_ * H_;
    k_cvt_bf16<<<(unsigned)((n + 255) / 256), blk256, 0, stream>>>(item, itemb, n);
  }
  k_w_to_bf16T<<<H_, blk128, 0, stream>>>(w1, wt1);
  k_w_to_bf16T<<<H_, blk128, 0, stream>>>(w3, wt3);
  k_w_to_bf16T<<<H_, blk128, 0, stream>>>(w4, wt4);
  k_w_to_bf16T<<<H_, blk128, 0, stream>>>(w6, wt6);
  k_pmean<<<1, blk128, 0, stream>>>(proto, pmean, accum);
  k_gram<<<L_, blk128, 0, stream>>>(proto, pmean, accum);
  k_fin<<<1, 64, 0, stream>>>(accum, out_scalar);

  // main pipeline (stream-ordered dependencies)
  k_gather<<<BS_, blk128, 0, stream>>>(item, pos, seq, ie, emb, mask);

  gemm_h128<0, 1><<<mstrips / 8, blk256, 0, stream>>>(ie, wt1, w2, nullptr, nullptr,
                                                      sa, mstrips);
  k_vcv<<<B_, blk128, 0, stream>>>(sa, mask, ie, vcv);
  k_protoscore<<<B_, blk128, 0, stream>>>(vcv, proto, sprot);
  k_topk<<<B_, blk128, 0, stream>>>(sprot, proto, ln2g, ln2b, cp, cpln);

  gemm_h128<1, 1><<<mstrips / 8, blk256, 0, stream>>>(ie, wt3, nullptr, ln1g, ln1b,
                                                      q, mstrips);
  k_pkt<<<B_, blk128, 0, stream>>>(q, cpln, mask, pkt);

  gemm_h128<0, 4><<<mstrips / 8, blk256, 0, stream>>>(emb, wt4, w5, nullptr, nullptr,
                                                      aksc, mstrips);
  k_ak<<<B_ * KP_, blk128, 0, stream>>>(aksc, mask, ak);
  k_interesting<<<B_ * KP_, blk128, 0, stream>>>(pkt, ak, emb, ln3g, ln3b, inter);

  k_xu<<<B_, blk128, 0, stream>>>(pkt, cp, xu);
  gemm_h128<0, 1><<<mstrips / 8, blk256, 0, stream>>>(xu, wt6, w7, nullptr, nullptr,
                                                      csc, mstrips);
  k_capt<<<B_, blk128, 0, stream>>>(csc, mask, xu, ln4g, ln4b, capt);
  k_final<<<B_, blk128, 0, stream>>>(capt, inter, finalf, finalb);

  // the big one: 1024 x 100000 scores (NT stores keep item_emb L2-resident)
  dim3 grid_sc((NTILES_ + 7) / 8, B_ / 16);     // 782 x 64 blocks, 8 waves/block
  gemm_scores<<<grid_sc, blk256, 0, stream>>>(finalb, itemb, scores);
}
